// GeometryAttentionLayer_4105988735893
// MI455X (gfx1250) — matrix operand order repaired and checked by reading, not compiled, not run on hardware
//
#include <hip/hip_runtime.h>
#include <hip/hip_bf16.h>

typedef __attribute__((ext_vector_type(16))) _Float16 v16h;
typedef __attribute__((ext_vector_type(8)))  float    v8f;

#define DEVI __device__ __forceinline__

// Async global->LDS path (gfx1250): guarded so a missing builtin falls back
// to plain vector copies instead of breaking the build.
#if defined(__has_builtin)
#if __has_builtin(__builtin_amdgcn_global_load_async_to_lds_b128)
#define USE_ASYNC_LDS 1
#endif
#endif

#ifdef USE_ASYNC_LDS
// Parameter types per hipcc diagnostic:
//   arg0: int __attribute__((vector_size(16))) __device__ * (global, AS1)
//   arg1: same pointee in LDS (AS3)
typedef int i32x4 __attribute__((vector_size(4 * sizeof(int))));
typedef __attribute__((address_space(1))) i32x4 glb_i32x4;
typedef __attribute__((address_space(3))) i32x4 lds_i32x4;
#endif

// ---------------------------------------------------------------------------
// WMMA fragment loaders (CDNA5 16x16x32 f16 layouts, wave32)
// ---------------------------------------------------------------------------

// A (16xK tile, row major). lanes 0-15: M=lane, elems 0..15 -> K=k0..k0+15;
// lanes 16-31: M=lane-16, K=k0+16..k0+31.
DEVI v16h load_a_frag(const _Float16* A, int lda, int m0, int k0, int lane) {
  const _Float16* p = A + (size_t)(m0 + (lane & 15)) * lda + k0 + ((lane >> 4) << 4);
  v16h a;
#pragma unroll
  for (int e = 0; e < 16; ++e) a[e] = p[e];
  return a;
}

// B fragment where B[k][n] = W[n][k]  (W row-major N x K).
DEVI v16h load_b_frag_t(const _Float16* W, int ldw, int n0, int k0, int lane) {
  const _Float16* p = W + (size_t)(n0 + (lane & 15)) * ldw + k0 + ((lane >> 4) << 4);
  v16h b;
#pragma unroll
  for (int e = 0; e < 16; ++e) b[e] = p[e];
  return b;
}

// B fragment where B[k][n] = V[k][n]  (V row-major K x N).
DEVI v16h load_b_frag_n(const _Float16* V, int ldv, int n0, int k0, int lane) {
  int n = n0 + (lane & 15);
  int kb = k0 + ((lane >> 4) << 4);
  v16h b;
#pragma unroll
  for (int e = 0; e < 16; ++e) b[e] = V[(size_t)(kb + e) * ldv + n];
  return b;
}

DEVI v8f wmma16(v16h a, v16h b, v8f c) {
  return __builtin_amdgcn_wmma_f32_16x16x32_f16(false, a, false, b, (short)0, c,
                                                false, false);
}

// ---------------------------------------------------------------------------
// f32 -> f16 conversion
// ---------------------------------------------------------------------------
__global__ void cvt_f32_to_f16_kernel(const float* __restrict__ in,
                                      _Float16* __restrict__ out, int n) {
  int i = blockIdx.x * blockDim.x + threadIdx.x;
  if (i < n) out[i] = (_Float16)in[i];
}

// ---------------------------------------------------------------------------
// Generic y = A @ W^T GEMM.  One wave = one 32x32 C tile (2x2 WMMA frags,
// A/B fragments each reused twice).  A is f16 [M x K], optionally split at
// splitK between A0/A1 (for concat).  W is f16 [N x K] row major.
// Epilogues: 0 store f32 | 1 store f16 | 2 relu f16 | 3 rotary (pe) f16
// ---------------------------------------------------------------------------
__global__ void gemm_wt_kernel(const _Float16* __restrict__ A0,
                               const _Float16* __restrict__ A1, int splitK,
                               const _Float16* __restrict__ W,
                               const float* __restrict__ pe,
                               float* __restrict__ outF,
                               _Float16* __restrict__ outH,
                               int M, int N, int K, int epi) {
  int tilesN = N >> 5;
  int tilesM = M >> 5;
  int wave = blockIdx.x * (blockDim.x >> 5) + (threadIdx.x >> 5);
  if (wave >= tilesM * tilesN) return;
  int lane = threadIdx.x & 31;
  int m0 = (wave / tilesN) << 5;
  int n0 = (wave % tilesN) << 5;

  v8f z = {};
  v8f c00 = z, c01 = z, c10 = z, c11 = z;
  for (int k0 = 0; k0 < K; k0 += 32) {
    const _Float16* As;
    int kk, lda;
    if (k0 < splitK) { As = A0; kk = k0;          lda = splitK; }
    else             { As = A1; kk = k0 - splitK; lda = K - splitK; }
    v16h a0 = load_a_frag(As, lda, m0, kk, lane);
    v16h a1 = load_a_frag(As, lda, m0 + 16, kk, lane);
    v16h b0 = load_b_frag_t(W, K, n0, k0, lane);
    v16h b1 = load_b_frag_t(W, K, n0 + 16, k0, lane);
    c00 = wmma16(a0, b0, c00);
    c01 = wmma16(a0, b1, c01);
    c10 = wmma16(a1, b0, c10);
    c11 = wmma16(a1, b1, c11);
  }

  v8f cf[2][2] = {{c00, c01}, {c10, c11}};
  int rhi = (lane >> 4) << 3;
  int lcol = lane & 15;
#pragma unroll
  for (int mi = 0; mi < 2; ++mi) {
#pragma unroll
    for (int ni = 0; ni < 2; ++ni) {
      int col = n0 + ni * 16 + lcol;
#pragma unroll
      for (int r = 0; r < 8; ++r) {
        int row = m0 + mi * 16 + r + rhi;
        float v = cf[mi][ni][r];
        if (epi == 3) {
          // interleaved rotary: even ch -> -x[ch+1]*sin ; odd ch -> +x[ch-1]*sin
          float partner = __shfl_xor(v, 1, 32);
          float x2 = (col & 1) ? partner : -partner;
          size_t pidx = ((size_t)row * N + col) * 2;
          v = v * pe[pidx] + x2 * pe[pidx + 1];
        } else if (epi == 2) {
          v = v > 0.f ? v : 0.f;
        }
        if (epi == 0) outF[(size_t)row * N + col] = v;
        else          outH[(size_t)row * N + col] = (_Float16)v;
      }
    }
  }
}

// ---------------------------------------------------------------------------
// Flash attention.  grid = (L/64, NHEAD), block = 128 (4 waves).
// Each wave owns 16 query rows of one head (head_dim 64).
// K/V tiles of 64 source rows staged in LDS via async global->LDS copies.
// ---------------------------------------------------------------------------
#define HD 64
#define PLD 40   // padded lda for P staging (16 x 32 tile)

__global__ void flash_attn_kernel(const _Float16* __restrict__ Q,
                                  const _Float16* __restrict__ Kg,
                                  const _Float16* __restrict__ Vg,
                                  const bool* __restrict__ xmask,
                                  const bool* __restrict__ smask,
                                  _Float16* __restrict__ O,
                                  int L, int S) {
  __shared__ _Float16 Kt[64 * HD];
  __shared__ _Float16 Vt[64 * HD];
  __shared__ _Float16 Pb[4][16 * PLD];

  const int head = blockIdx.y;
  const int hoff = head * HD;
  const int wid = threadIdx.x >> 5;
  const int lane = threadIdx.x & 31;
  const int m0 = blockIdx.x * 64 + wid * 16;
  const int rhi = (lane >> 4) << 3;
  const float NEGBIG = -1e30f;

  v16h qa0 = load_a_frag(Q + hoff, 256, m0, 0, lane);
  v16h qa1 = load_a_frag(Q + hoff, 256, m0, 32, lane);

  v8f zero8 = {};
  v8f ov[4] = {zero8, zero8, zero8, zero8};
  float mrow[8], lrow[8];
  bool xmr[8];
#pragma unroll
  for (int r = 0; r < 8; ++r) {
    mrow[r] = NEGBIG;
    lrow[r] = 0.f;
    xmr[r] = xmask[m0 + r + rhi];
  }

  for (int s0 = 0; s0 < S; s0 += 64) {
    __syncthreads();  // previous tile fully consumed by all waves
    // ---- K/V tile staging: 64 rows x 64 halves, 16B granules
    for (int t = threadIdx.x; t < 512; t += blockDim.x) {
      int srow = t >> 3;
      int dcol = (t & 7) << 3;
      const _Float16* gk = &Kg[(size_t)(s0 + srow) * 256 + hoff + dcol];
      const _Float16* gv = &Vg[(size_t)(s0 + srow) * 256 + hoff + dcol];
#ifdef USE_ASYNC_LDS
      __builtin_amdgcn_global_load_async_to_lds_b128(
          (glb_i32x4*)gk, (lds_i32x4*)&Kt[srow * HD + dcol], 0, 0);
      __builtin_amdgcn_global_load_async_to_lds_b128(
          (glb_i32x4*)gv, (lds_i32x4*)&Vt[srow * HD + dcol], 0, 0);
#else
      *(float4*)(&Kt[srow * HD + dcol]) = *(const float4*)gk;
      *(float4*)(&Vt[srow * HD + dcol]) = *(const float4*)gv;
#endif
    }
#ifdef USE_ASYNC_LDS
#if __has_builtin(__builtin_amdgcn_s_wait_asynccnt)
    __builtin_amdgcn_s_wait_asynccnt(0);
#else
    asm volatile("s_wait_asynccnt 0x0" ::: "memory");
#endif
#endif
    if (s0 + 64 < S) {
      __builtin_prefetch(&Kg[(size_t)(s0 + 64 + (threadIdx.x >> 1)) * 256 + hoff], 0, 1);
      __builtin_prefetch(&Vg[(size_t)(s0 + 64 + (threadIdx.x >> 1)) * 256 + hoff], 0, 1);
    }
    __syncthreads();

    // ---- scores S[16 x 64] in four 16x16 fragments
    v8f sc[4];
#pragma unroll
    for (int j = 0; j < 4; ++j) {
      v8f s = {};
      v16h b0 = load_b_frag_t(Kt, HD, j * 16, 0, lane);   // B[k=d][n=s] = K[s][d]
      v16h b1 = load_b_frag_t(Kt, HD, j * 16, 32, lane);
      s = wmma16(qa0, b0, s);
      s = wmma16(qa1, b1, s);
      sc[j] = s;
    }

    // ---- scale by 1/sqrt(64), then mask (x_mask & ~source_mask -> -inf)
#pragma unroll
    for (int j = 0; j < 4; ++j) {
      bool col_masked = !smask[s0 + j * 16 + (lane & 15)];
#pragma unroll
      for (int r = 0; r < 8; ++r) {
        float v = sc[j][r] * 0.125f;
        if (xmr[r] && col_masked) v = NEGBIG;
        sc[j][r] = v;
      }
    }

    // ---- streaming softmax
    float mnew[8], corr[8], rsum[8];
#pragma unroll
    for (int r = 0; r < 8; ++r) {
      float t = fmaxf(fmaxf(sc[0][r], sc[1][r]), fmaxf(sc[2][r], sc[3][r]));
      t = fmaxf(t, __shfl_xor(t, 1, 32));
      t = fmaxf(t, __shfl_xor(t, 2, 32));
      t = fmaxf(t, __shfl_xor(t, 4, 32));
      t = fmaxf(t, __shfl_xor(t, 8, 32));
      mnew[r] = fmaxf(mrow[r], t);
      corr[r] = __expf(mrow[r] - mnew[r]);
      rsum[r] = 0.f;
    }
#pragma unroll
    for (int j = 0; j < 4; ++j) {
#pragma unroll
      for (int r = 0; r < 8; ++r) {
        float v = sc[j][r];
        float p = __expf(v - mnew[r]);
        if (v == NEGBIG) p = 0.f;
        sc[j][r] = p;
        rsum[r] += p;
      }
    }
#pragma unroll
    for (int r = 0; r < 8; ++r) {
      float t = rsum[r];
      t += __shfl_xor(t, 1, 32);
      t += __shfl_xor(t, 2, 32);
      t += __shfl_xor(t, 4, 32);
      t += __shfl_xor(t, 8, 32);
      lrow[r] = lrow[r] * corr[r] + t;
      mrow[r] = mnew[r];
    }
#pragma unroll
    for (int f = 0; f < 4; ++f)
#pragma unroll
      for (int r = 0; r < 8; ++r) ov[f][r] *= corr[r];

    // ---- P @ V: bounce P (C layout) through LDS into A layout, 32-s chunks
    _Float16* pb = &Pb[wid][0];
#pragma unroll
    for (int half = 0; half < 2; ++half) {
#pragma unroll
      for (int jj = 0; jj < 2; ++jj) {
        int j = half * 2 + jj;
#pragma unroll
        for (int r = 0; r < 8; ++r)
          pb[(size_t)(r + rhi) * PLD + jj * 16 + (lane & 15)] = (_Float16)sc[j][r];
      }
      asm volatile("s_wait_dscnt 0x0" ::: "memory");  // wave-local store->load
      v16h pa = load_a_frag(pb, PLD, 0, 0, lane);
#pragma unroll
      for (int f = 0; f < 4; ++f) {
        v16h vb = load_b_frag_n(Vt, HD, f * 16, half * 32, lane);  // B[k=s][n=d]
        ov[f] = wmma16(pa, vb, ov[f]);
      }
    }
  }

  // ---- normalize and write O (f16, [L x 256])
#pragma unroll
  for (int f = 0; f < 4; ++f) {
#pragma unroll
    for (int r = 0; r < 8; ++r) {
      int row = m0 + r + rhi;
      float denom = lrow[r];
      float val = denom > 0.f ? ov[f][r] / denom : 0.f;
      O[(size_t)row * 256 + hoff + f * 16 + (lane & 15)] = (_Float16)val;
    }
  }
}

// ---------------------------------------------------------------------------
// LayerNorm over last dim (D=256). grid = rows, block = 256.
// ---------------------------------------------------------------------------
__global__ void layernorm_kernel(const float* __restrict__ in,
                                 const float* __restrict__ gamma,
                                 const float* __restrict__ beta,
                                 const float* __restrict__ resid,
                                 float* __restrict__ outF,
                                 _Float16* __restrict__ outH) {
  const int D = 256;
  int row = blockIdx.x;
  int c = threadIdx.x;
  __shared__ float red[256];

  float v = in[(size_t)row * D + c];
  red[c] = v;
  __syncthreads();
  for (int off = 128; off > 0; off >>= 1) {
    if (c < off) red[c] += red[c + off];
    __syncthreads();
  }
  float mu = red[0] * (1.0f / D);
  __syncthreads();
  float d = v - mu;
  red[c] = d * d;
  __syncthreads();
  for (int off = 128; off > 0; off >>= 1) {
    if (c < off) red[c] += red[c + off];
    __syncthreads();
  }
  float var = red[0] * (1.0f / D);
  float y = d * rsqrtf(var + 1e-5f) * gamma[c] + beta[c];
  if (outH) outH[(size_t)row * D + c] = (_Float16)y;
  if (outF) outF[(size_t)row * D + c] =
      (resid ? resid[(size_t)row * D + c] : 0.f) + y;
}

// ---------------------------------------------------------------------------
// Host-side orchestration
// ---------------------------------------------------------------------------
extern "C" void kernel_launch(void* const* d_in, const int* in_sizes, int n_in,
                              void* d_out, int out_size, void* d_ws, size_t ws_size,
                              hipStream_t stream) {
  const float* x   = (const float*)d_in[0];
  const float* src = (const float*)d_in[1];
  const float* xpe = (const float*)d_in[2];
  const float* spe = (const float*)d_in[3];
  const bool*  xm  = (const bool*)d_in[4];
  const bool*  sm  = (const bool*)d_in[5];
  const float* Wq  = (const float*)d_in[6];
  const float* Wk  = (const float*)d_in[7];
  const float* Wv  = (const float*)d_in[8];
  const float* Wm  = (const float*)d_in[9];
  const float* W1  = (const float*)d_in[10];
  const float* W2  = (const float*)d_in[11];
  const float* g1  = (const float*)d_in[12];
  const float* b1  = (const float*)d_in[13];
  const float* g2  = (const float*)d_in[14];
  const float* b2  = (const float*)d_in[15];
  float* out = (float*)d_out;

  const int L = 4096, D = 256, D2 = 512;

  char* p = (char*)d_ws;
  auto alloc = [&](size_t bytes) -> char* {
    char* r = p;
    p += (bytes + 255) & ~(size_t)255;
    return r;
  };
  _Float16* x16   = (_Float16*)alloc((size_t)L * D * 2);
  _Float16* s16   = (_Float16*)alloc((size_t)L * D * 2);
  _Float16* Wq16  = (_Float16*)alloc((size_t)D * D * 2);
  _Float16* Wk16  = (_Float16*)alloc((size_t)D * D * 2);
  _Float16* Wv16  = (_Float16*)alloc((size_t)D * D * 2);
  _Float16* Wm16  = (_Float16*)alloc((size_t)D * D * 2);
  _Float16* W116  = (_Float16*)alloc((size_t)D2 * D2 * 2);
  _Float16* W216  = (_Float16*)alloc((size_t)D * D2 * 2);
  _Float16* Q16   = (_Float16*)alloc((size_t)L * D * 2);
  _Float16* K16   = (_Float16*)alloc((size_t)L * D * 2);
  _Float16* V16   = (_Float16*)alloc((size_t)L * D * 2);
  _Float16* O16   = (_Float16*)alloc((size_t)L * D * 2);
  float*    msgF  = (float*)   alloc((size_t)L * D * 4);
  _Float16* msg16 = (_Float16*)alloc((size_t)L * D * 2);
  _Float16* h16   = (_Float16*)alloc((size_t)L * D2 * 2);
  float*    y2F   = (float*)   alloc((size_t)L * D * 4);

  auto cvt = [&](const float* fin, _Float16* fout, int n) {
    cvt_f32_to_f16_kernel<<<(n + 255) / 256, 256, 0, stream>>>(fin, fout, n);
  };
  cvt(x,   x16,  L * D);
  cvt(src, s16,  L * D);
  cvt(Wq,  Wq16, D * D);
  cvt(Wk,  Wk16, D * D);
  cvt(Wv,  Wv16, D * D);
  cvt(Wm,  Wm16, D * D);
  cvt(W1,  W116, D2 * D2);
  cvt(W2,  W216, D * D2);

  // GEMMs: one 32x32 tile per wave, 8 waves per 256-thread block.
  auto gemm = [&](const _Float16* A0, const _Float16* A1, int splitK,
                  const _Float16* W, const float* pe, float* oF, _Float16* oH,
                  int M, int N, int K, int epi) {
    int tiles = (M >> 5) * (N >> 5);
    gemm_wt_kernel<<<(tiles + 7) / 8, 256, 0, stream>>>(A0, A1, splitK, W, pe,
                                                        oF, oH, M, N, K, epi);
  };
  gemm(x16, nullptr, D, Wq16, xpe, nullptr, Q16, L, D, D, 3);     // Q = rot(x@Wq^T)
  gemm(s16, nullptr, D, Wk16, spe, nullptr, K16, L, D, D, 3);     // K = rot(s@Wk^T)
  gemm(s16, nullptr, D, Wv16, nullptr, nullptr, V16, L, D, D, 1); // V = s@Wv^T

  // Flash attention: grid (L/64 row-blocks, 4 heads), 4 waves per block.
  flash_attn_kernel<<<dim3(L / 64, 4), 128, 0, stream>>>(Q16, K16, V16, xm, sm,
                                                         O16, L, L);

  // message = LN(O @ Wm^T)
  gemm(O16, nullptr, D, Wm16, nullptr, msgF, nullptr, L, D, D, 0);
  layernorm_kernel<<<L, 256, 0, stream>>>(msgF, g1, b1, nullptr, nullptr, msg16);

  // h = relu(concat(x, message) @ W1^T)
  gemm(x16, msg16, D, W116, nullptr, nullptr, h16, L, D2, D2, 2);

  // y2 = h @ W2^T ; out = x + LN(y2)
  gemm(h16, nullptr, D2, W216, nullptr, y2F, nullptr, L, D, D2, 0);
  layernorm_kernel<<<L, 256, 0, stream>>>(y2F, g2, b2, x, out, nullptr);
}
